// MultiHeadAttention_9526237462769
// MI455X (gfx1250) — matrix operand order repaired and checked by reading, not compiled, not run on hardware
//
#include <hip/hip_runtime.h>
#include <math.h>

// MI455X / gfx1250 — bf16 WMMA flash-attention MHA pipeline.
// All matmuls use v_wmma_f32_16x16x32_bf16 (fp32 accumulate).
// K-loops are software-pipelined (double-buffered fragments) so WMMAs
// overlap with in-flight global_load_b128 instead of s_wait_loadcnt 0.

typedef __bf16 bf16;
typedef __attribute__((ext_vector_type(16))) __bf16 v16bf;
typedef __attribute__((ext_vector_type(8)))  __bf16 v8bf;
typedef __attribute__((ext_vector_type(8)))  float  v8f;

#define BB 32
#define SS 512
#define DD 768
#define HH 12
#define EE 64

__device__ __forceinline__ bf16 f2bf(float f) {
  union { float f; unsigned u; } x; x.f = f;
  unsigned r = x.u + 0x7FFFu + ((x.u >> 16) & 1u);   // round-to-nearest-even
  unsigned short h = (unsigned short)(r >> 16);
  return __builtin_bit_cast(bf16, h);
}

__device__ __forceinline__ v8bf ld8(const bf16* p) {
  return *reinterpret_cast<const v8bf*>(p);
}
__device__ __forceinline__ v16bf cat(v8bf lo, v8bf hi) {
  return __builtin_shufflevector(lo, hi, 0,1,2,3,4,5,6,7,8,9,10,11,12,13,14,15);
}
__device__ __forceinline__ v8f wmma_bf16(v16bf a, v16bf b, v8f c) {
  return __builtin_amdgcn_wmma_f32_16x16x32_bf16(
      /*neg_a=*/false, a, /*neg_b=*/false, b,
      /*c_mod=*/(short)0, c, /*reuse_a=*/false, /*reuse_b=*/false);
}

// A-fragment (16x32, M=lane%16 rows): K interleave = j + 8*hi (j<8), j+8+8*hi (j>=8)
__device__ __forceinline__ v16bf loadA(const bf16* row, int k0, int hi) {
  return cat(ld8(row + k0 + 8*hi), ld8(row + k0 + 16 + 8*hi));
}
// B-fragment (32x16, N=lane%16 cols, K-major rows): K interleave = j + 8*hi (j<8), j+8+8*hi (j>=8)
__device__ __forceinline__ v16bf loadB(const bf16* row, int k0, int hi) {
  return cat(ld8(row + k0 + 8*hi), ld8(row + k0 + 16 + 8*hi));
}

// ---------------- conversion kernels ----------------
__global__ void __launch_bounds__(256) cvt_kernel(const float* __restrict__ in,
                                                  bf16* __restrict__ out, int n) {
  int i = blockIdx.x * blockDim.x + threadIdx.x;
  if (i < n) out[i] = f2bf(in[i]);
}

// out[h][e][d] = in[h][d][e]  (store weights K(=d)-major for contiguous B-fragments)
__global__ void __launch_bounds__(256) cvt_wT_kernel(const float* __restrict__ in,
                                                     bf16* __restrict__ out,
                                                     int Hn, int Dn, int En) {
  int i = blockIdx.x * blockDim.x + threadIdx.x;
  int n = Hn * Dn * En;
  if (i < n) {
    int h   = i / (En * Dn);
    int rem = i - h * En * Dn;
    int e   = rem / Dn;
    int d   = rem - e * Dn;
    out[i] = f2bf(in[((size_t)h * Dn + d) * En + e]);
  }
}

// ---------------- fused QKV projection ----------------
// One wave -> one 16x16 tile of q, k, v for a given (b,h,s-tile,e-tile).
// q,k stored [B,H,S,E] bf16; v stored transposed [B,H,E,S] bf16.
__global__ void __launch_bounds__(256) qkv_kernel(
    const bf16* __restrict__ xb,
    const bf16* __restrict__ wqT, const bf16* __restrict__ wkT, const bf16* __restrict__ wvT,
    const float* __restrict__ bq, const float* __restrict__ bk, const float* __restrict__ bv,
    bf16* __restrict__ q, bf16* __restrict__ k, bf16* __restrict__ vT)
{
  int wid  = blockIdx.x * (blockDim.x >> 5) + (threadIdx.x >> 5);
  int lane = threadIdx.x & 31;
  int hi = lane >> 4, lr = lane & 15;
  int et = wid & 3;
  int st = (wid >> 2) & 31;
  int h  = (wid >> 7) % HH;
  int b  = wid / (4 * 32 * HH);

  const int e = et * 16 + lr;
  const bf16* arow  = xb  + (size_t)(b * SS + st * 16 + lr) * DD;
  const bf16* bqrow = wqT + (size_t)(h * EE + e) * DD;
  const bf16* bkrow = wkT + (size_t)(h * EE + e) * DD;
  const bf16* bvrow = wvT + (size_t)(h * EE + e) * DD;

  v8f cq = {}, ck = {}, cv = {};
  // software pipeline: fragments for step d0 are loaded during step d0-32
  v16bf a  = loadA(arow,  0, hi);
  v16bf fq = loadB(bqrow, 0, hi);
  v16bf fk = loadB(bkrow, 0, hi);
  v16bf fv = loadB(bvrow, 0, hi);
  #pragma unroll 2
  for (int d0 = 0; d0 < DD - 32; d0 += 32) {
    v16bf a2  = loadA(arow,  d0 + 32, hi);
    v16bf fq2 = loadB(bqrow, d0 + 32, hi);
    v16bf fk2 = loadB(bkrow, d0 + 32, hi);
    v16bf fv2 = loadB(bvrow, d0 + 32, hi);
    cq = wmma_bf16(a, fq, cq);
    ck = wmma_bf16(a, fk, ck);
    cv = wmma_bf16(a, fv, cv);
    a = a2; fq = fq2; fk = fk2; fv = fv2;
  }
  cq = wmma_bf16(a, fq, cq);
  ck = wmma_bf16(a, fk, ck);
  cv = wmma_bf16(a, fv, cv);

  float biq = bq[h * EE + e], bik = bk[h * EE + e], biv = bv[h * EE + e];
  size_t bh = (size_t)(b * HH + h);
  #pragma unroll
  for (int r = 0; r < 8; ++r) {
    int srow = st * 16 + r + 8 * hi;
    q[(bh * SS + srow) * EE + e] = f2bf(cq[r] + biq);
    k[(bh * SS + srow) * EE + e] = f2bf(ck[r] + bik);
    vT[(bh * EE + e) * SS + srow] = f2bf(cv[r] + biv);
  }
}

// ---------------- flash attention ----------------
// One wave owns a 16-row s-block of one (b,h). Computes S^T = K·Q^T so the
// score registers line up directly with the A-fragment layout of P for P·V.
__global__ void __launch_bounds__(256) attn_kernel(
    const bf16* __restrict__ q, const bf16* __restrict__ k,
    const bf16* __restrict__ vT, bf16* __restrict__ ao)
{
  int wid  = blockIdx.x * (blockDim.x >> 5) + (threadIdx.x >> 5);
  int lane = threadIdx.x & 31;
  int hi = lane >> 4, lr = lane & 15;
  int st = wid & 31;
  int h  = (wid >> 5) % HH;
  int b  = wid / (32 * HH);

  size_t bh = (size_t)(b * HH + h);
  const bf16* qrow = q + (bh * SS + st * 16 + lr) * EE;
  v16bf qB0 = loadB(qrow, 0, hi);    // e = 0..31  (Q is the B operand of K·Q^T)
  v16bf qB1 = loadB(qrow, 32, hi);   // e = 32..63

  const bf16* krow = k  + (bh * SS + lr) * EE;   // + tc*EE per chunk
  const bf16* vrow = vT + (bh * EE + lr) * SS;   // + tc + 8*hi per chunk

  v8f O0 = {}, O1 = {}, O2 = {}, O3 = {};
  float m = -__builtin_inff();
  float l = 0.f;
  const float scale = 0.125f;  // 1/sqrt(64)

  // pipeline: K-tile fragments for chunk tc are loaded during chunk tc-32
  v16bf k00 = loadA(krow, 0, hi),           k01 = loadA(krow, 32, hi);
  v16bf k10 = loadA(krow + 16 * EE, 0, hi), k11 = loadA(krow + 16 * EE, 32, hi);

  for (int tc = 0; tc < SS; tc += 32) {
    // V loads issued early: in flight through the whole softmax section
    const bf16* vbase = vrow + tc + 8 * hi;
    v16bf vB0 = cat(ld8(vbase),           ld8(vbase + 16));
    v16bf vB1 = cat(ld8(vbase + 16 * SS), ld8(vbase + 16 * SS + 16));
    v16bf vB2 = cat(ld8(vbase + 32 * SS), ld8(vbase + 32 * SS + 16));
    v16bf vB3 = cat(ld8(vbase + 48 * SS), ld8(vbase + 48 * SS + 16));

    v8f sa = {}; sa = wmma_bf16(k00, qB0, sa); sa = wmma_bf16(k01, qB1, sa);
    v8f sb = {}; sb = wmma_bf16(k10, qB0, sb); sb = wmma_bf16(k11, qB1, sb);

    if (tc + 32 < SS) {  // prefetch next K-tile while softmax runs
      const bf16* kn = krow + (size_t)(tc + 32) * EE;
      k00 = loadA(kn, 0, hi);           k01 = loadA(kn, 32, hi);
      k10 = loadA(kn + 16 * EE, 0, hi); k11 = loadA(kn + 16 * EE, 32, hi);
    }

    // per-s-row stats: registers hold t = r + 8*hi for s = lane%16
    float mloc = -__builtin_inff();
    #pragma unroll
    for (int r = 0; r < 8; ++r) {
      sa[r] *= scale; sb[r] *= scale;
      mloc = fmaxf(mloc, fmaxf(sa[r], sb[r]));
    }
    mloc = fmaxf(mloc, __shfl_xor(mloc, 16, 32));
    float mnew  = fmaxf(m, mloc);
    float alpha = __expf(m - mnew);     // 0 on first chunk
    float lloc  = 0.f;
    v16bf pA;
    #pragma unroll
    for (int r = 0; r < 8; ++r) {
      float pa = __expf(sa[r] - mnew);
      float pb = __expf(sb[r] - mnew);
      lloc += pa + pb;
      pA[r]     = f2bf(pa);   // A-frag K = r + 8*hi       (t-tile a)
      pA[r + 8] = f2bf(pb);   // A-frag K = 16 + r + 8*hi  (t-tile b)
    }
    lloc += __shfl_xor(lloc, 16, 32);
    l = l * alpha + lloc;
    m = mnew;

    // rescale O: its rows are s = r + 8*hi; alpha lives at lane s (0..15)
    #pragma unroll
    for (int r = 0; r < 8; ++r) {
      float aO = __shfl(alpha, r + 8 * hi, 32);
      O0[r] *= aO; O1[r] *= aO; O2[r] *= aO; O3[r] *= aO;
    }

    O0 = wmma_bf16(pA, vB0, O0);
    O1 = wmma_bf16(pA, vB1, O1);
    O2 = wmma_bf16(pA, vB2, O2);
    O3 = wmma_bf16(pA, vB3, O3);
  }

  // normalize and store head output into [B,S,H*E] bf16 (ready for out-proj A)
  #pragma unroll
  for (int r = 0; r < 8; ++r) {
    float lrow = __shfl(l, r + 8 * hi, 32);
    float inv  = 1.0f / lrow;
    int srow = st * 16 + r + 8 * hi;
    bf16* dst = ao + (size_t)(b * SS + srow) * DD + h * EE;
    dst[0  + lr] = f2bf(O0[r] * inv);
    dst[16 + lr] = f2bf(O1[r] * inv);
    dst[32 + lr] = f2bf(O2[r] * inv);
    dst[48 + lr] = f2bf(O3[r] * inv);
  }
}

// ---------------- output projection ----------------
__global__ void __launch_bounds__(256) proj_kernel(
    const bf16* __restrict__ ao, const bf16* __restrict__ woT,
    const float* __restrict__ bo, float* __restrict__ out)
{
  int wid  = blockIdx.x * (blockDim.x >> 5) + (threadIdx.x >> 5);
  int lane = threadIdx.x & 31;
  int hi = lane >> 4, lr = lane & 15;
  int nt = wid % (DD / 16);
  int mt = wid / (DD / 16);

  const bf16* arow = ao  + (size_t)(mt * 16 + lr) * DD;
  const bf16* brow = woT + (size_t)(nt * 16 + lr) * DD;
  v8f c = {};
  v16bf a = loadA(arow, 0, hi);
  v16bf w = loadB(brow, 0, hi);
  #pragma unroll 2
  for (int d0 = 0; d0 < DD - 32; d0 += 32) {
    v16bf a2 = loadA(arow, d0 + 32, hi);
    v16bf w2 = loadB(brow, d0 + 32, hi);
    c = wmma_bf16(a, w, c);
    a = a2; w = w2;
  }
  c = wmma_bf16(a, w, c);

  float bias = bo[nt * 16 + lr];
  #pragma unroll
  for (int r = 0; r < 8; ++r)
    out[(size_t)(mt * 16 + r + 8 * hi) * DD + nt * 16 + lr] = c[r] + bias;
}

extern "C" void kernel_launch(void* const* d_in, const int* in_sizes, int n_in,
                              void* d_out, int out_size, void* d_ws, size_t ws_size,
                              hipStream_t stream) {
  (void)in_sizes; (void)n_in; (void)out_size; (void)ws_size;
  const float* x  = (const float*)d_in[0];
  const float* Wq = (const float*)d_in[1];
  const float* Wk = (const float*)d_in[2];
  const float* Wv = (const float*)d_in[3];
  const float* bq = (const float*)d_in[4];
  const float* bk = (const float*)d_in[5];
  const float* bv = (const float*)d_in[6];
  const float* Wo = (const float*)d_in[7];
  const float* bo = (const float*)d_in[8];

  char* base = (char*)d_ws;
  size_t off = 0;
  auto alloc = [&](size_t bytes) -> char* {
    char* p = base + off;
    off = (off + bytes + 255) & ~(size_t)255;
    return p;
  };
  bf16* xb  = (bf16*)alloc((size_t)BB * SS * DD * 2);  // x in bf16
  bf16* wqT = (bf16*)alloc((size_t)HH * EE * DD * 2);  // [H,E,D]
  bf16* wkT = (bf16*)alloc((size_t)HH * EE * DD * 2);
  bf16* wvT = (bf16*)alloc((size_t)HH * EE * DD * 2);
  bf16* woT = (bf16*)alloc((size_t)DD * DD * 2);       // [out,in]
  bf16* qb  = (bf16*)alloc((size_t)BB * HH * SS * EE * 2);  // [B,H,S,E]
  bf16* kb  = (bf16*)alloc((size_t)BB * HH * SS * EE * 2);  // [B,H,S,E]
  bf16* vTb = (bf16*)alloc((size_t)BB * HH * EE * SS * 2);  // [B,H,E,S]
  bf16* aob = (bf16*)alloc((size_t)BB * SS * DD * 2);       // [B,S,H*E]

  const int nx = BB * SS * DD;
  cvt_kernel<<<(nx + 255) / 256, 256, 0, stream>>>(x, xb, nx);
  const int nw = HH * DD * EE;
  cvt_wT_kernel<<<(nw + 255) / 256, 256, 0, stream>>>(Wq, wqT, HH, DD, EE);
  cvt_wT_kernel<<<(nw + 255) / 256, 256, 0, stream>>>(Wk, wkT, HH, DD, EE);
  cvt_wT_kernel<<<(nw + 255) / 256, 256, 0, stream>>>(Wv, wvT, HH, DD, EE);
  const int nwo = DD * DD;
  cvt_wT_kernel<<<(nwo + 255) / 256, 256, 0, stream>>>(Wo, woT, 1, DD, DD);

  // QKV: B*H * (S/16) * (E/16) = 49152 wave-tiles, 8 waves/block
  qkv_kernel<<<49152 / 8, 256, 0, stream>>>(xb, wqT, wkT, wvT, bq, bk, bv, qb, kb, vTb);
  // Attention: B*H*(S/16) = 12288 wave-tasks
  attn_kernel<<<12288 / 8, 256, 0, stream>>>(qb, kb, vTb, aob);
  // Output projection: (B*S/16)*(D/16) = 49152 wave-tiles
  proj_kernel<<<49152 / 8, 256, 0, stream>>>(aob, woT, bo, (float*)d_out);
}
